// GAT_33079838114679
// MI455X (gfx1250) — compile-verified, matrix operand order checked by reading, not run on hardware
//
#include <hip/hip_runtime.h>
#include <hip/hip_bf16.h>

#define IN_DIM   256
#define HIDDEN   64
#define HEADS1   4
#define D1       256   // HEADS1*HIDDEN
#define NCLS     64
#define NEG_SLOPE 0.2f

typedef __attribute__((ext_vector_type(16))) __bf16 bf16x16;
typedef __attribute__((ext_vector_type(8)))  __bf16 bf16x8;
typedef __attribute__((ext_vector_type(8)))  float  f32x8;

// ---------------- helpers ----------------

__device__ __forceinline__ unsigned fkey(float f) {
  unsigned u = __float_as_uint(f);
  return (u & 0x80000000u) ? ~u : (u | 0x80000000u);
}
__device__ __forceinline__ float funkey(unsigned u) {
  u = (u & 0x80000000u) ? (u & 0x7fffffffu) : ~u;
  return __uint_as_float(u);
}
__device__ __forceinline__ float lrelu(float v) { return v > 0.0f ? v : NEG_SLOPE * v; }

// Load 16 bf16 per lane in the CDNA5 16-bit A/B fragment layout.
// Caller passes p already offset by (lane>=16 ? 8 : 0): elements 0..7 = K+0..7,
// elements 8..15 = K+16..23 (lanes 0-15) / K+24..31 (lanes 16-31).
__device__ __forceinline__ bf16x16 load_frag(const __bf16* p) {
  bf16x8 lo = *(const bf16x8*)(p);
  bf16x8 hi = *(const bf16x8*)(p + 16);
  return __builtin_shufflevector(lo, hi, 0,1,2,3,4,5,6,7,8,9,10,11,12,13,14,15);
}

// ---------------- conversion kernels ----------------

__global__ void k_f32_to_bf16(const float* __restrict__ in, __bf16* __restrict__ out, long n) {
  long i = (long)blockIdx.x * blockDim.x + threadIdx.x;
  if (i < n) out[i] = (__bf16)in[i];
}

// in: rows x cols (f32, row-major)  ->  out: cols x rows (bf16) so each output
// row (a GEMM B-column) is contiguous in K.
__global__ void k_transpose_bf16(const float* __restrict__ in, __bf16* __restrict__ out,
                                 int rows, int cols) {
  int i = blockIdx.x * blockDim.x + threadIdx.x;
  if (i < rows * cols) {
    int r = i / cols, c = i % cols;
    out[(size_t)c * rows + r] = (__bf16)in[i];
  }
}

// ---------------- WMMA GEMM kernels ----------------
// C[M,N] = A[M,K] (bf16, row-major) * BT[N,K]^T (bf16, K-contig per column).
// One wave -> one 16x16 tile; K stepped by 32 via v_wmma_f32_16x16x32_bf16.

__global__ void k_wmma_gemm(const __bf16* __restrict__ A, const __bf16* __restrict__ BT,
                            float* __restrict__ C, int M, int N, int K) {
  int wave   = (blockIdx.x * blockDim.x + threadIdx.x) >> 5;
  int tilesN = N >> 4;
  int total  = (M >> 4) * tilesN;
  if (wave >= total) return;                  // wave-uniform: EXEC stays all-ones
  int tm = wave / tilesN, tn = wave % tilesN;
  int lane = threadIdx.x & 31;
  int half = lane >> 4, lr = lane & 15;

  const __bf16* ap = A  + (size_t)(tm * 16 + lr) * K + half * 8;
  const __bf16* bp = BT + (size_t)(tn * 16 + lr) * K + half * 8;

  f32x8 acc = {};
  for (int kb = 0; kb < K; kb += 32) {
    bf16x16 a = load_frag(ap + kb);
    bf16x16 b = load_frag(bp + kb);
    acc = __builtin_amdgcn_wmma_f32_16x16x32_bf16(false, a, false, b,
                                                  (short)0, acc, false, false);
  }
  // C/D layout: VGPR i -> row (half*8 + i), col lr
  float* cp = C + (size_t)(tm * 16 + half * 8) * N + tn * 16 + lr;
#pragma unroll
  for (int i = 0; i < 8; ++i) cp[(size_t)i * N] = acc[i];
}

// Dual-B variant: two GEMMs sharing the A fragment (layer-2 main + residual).
__global__ void k_wmma_gemm_dual(const __bf16* __restrict__ A,
                                 const __bf16* __restrict__ B1T,
                                 const __bf16* __restrict__ B2T,
                                 float* __restrict__ C1, float* __restrict__ C2,
                                 int M, int N, int K) {
  int wave   = (blockIdx.x * blockDim.x + threadIdx.x) >> 5;
  int tilesN = N >> 4;
  int total  = (M >> 4) * tilesN;
  if (wave >= total) return;
  int tm = wave / tilesN, tn = wave % tilesN;
  int lane = threadIdx.x & 31;
  int half = lane >> 4, lr = lane & 15;

  const __bf16* ap  = A   + (size_t)(tm * 16 + lr) * K + half * 8;
  const __bf16* b1p = B1T + (size_t)(tn * 16 + lr) * K + half * 8;
  const __bf16* b2p = B2T + (size_t)(tn * 16 + lr) * K + half * 8;

  f32x8 acc1 = {}, acc2 = {};
  for (int kb = 0; kb < K; kb += 32) {
    bf16x16 a  = load_frag(ap  + kb);
    bf16x16 b1 = load_frag(b1p + kb);
    bf16x16 b2 = load_frag(b2p + kb);
    acc1 = __builtin_amdgcn_wmma_f32_16x16x32_bf16(false, a, false, b1,
                                                   (short)0, acc1, false, false);
    acc2 = __builtin_amdgcn_wmma_f32_16x16x32_bf16(false, a, false, b2,
                                                   (short)0, acc2, false, false);
  }
  float* c1p = C1 + (size_t)(tm * 16 + half * 8) * N + tn * 16 + lr;
  float* c2p = C2 + (size_t)(tm * 16 + half * 8) * N + tn * 16 + lr;
#pragma unroll
  for (int i = 0; i < 8; ++i) { c1p[(size_t)i * N] = acc1[i]; c2p[(size_t)i * N] = acc2[i]; }
}

// ---------------- attention kernels ----------------

// el[n,h] = sum_d feat[n,h*D+d]*aL[h,d]; same for er.
__global__ void k_eler(const float* __restrict__ feat, const float* __restrict__ aL,
                       const float* __restrict__ aR, float* __restrict__ el,
                       float* __restrict__ er, int nN, int H, int D) {
  int i = blockIdx.x * blockDim.x + threadIdx.x;
  if (i >= nN * H) return;
  int n = i / H, h = i % H;
  const float* f = feat + (size_t)n * H * D + h * D;
  const float* l = aL + h * D;
  const float* r = aR + h * D;
  float sl = 0.0f, sr = 0.0f;
  for (int d = 0; d < D; ++d) { float v = f[d]; sl += v * l[d]; sr += v * r[d]; }
  el[i] = sl; er[i] = sr;
}

// segment-max of leaky_relu(el[src]+er[dst]) over dst, via ordered-uint atomicMax
__global__ void k_edge_max(const int* __restrict__ src, const int* __restrict__ dst,
                           const float* __restrict__ el, const float* __restrict__ er,
                           unsigned* __restrict__ m, long nEH, int H) {
  long i = (long)blockIdx.x * blockDim.x + threadIdx.x;
  if (i >= nEH) return;
  int e = (int)(i / H), h = (int)(i % H);
  int s = src[e], d = dst[e];
  float v = lrelu(el[(size_t)s * H + h] + er[(size_t)d * H + h]);
  atomicMax(&m[(size_t)d * H + h], fkey(v));
}

// ex = exp(e - max); store per-edge and segment-sum into ssum[dst]
__global__ void k_edge_exp(const int* __restrict__ src, const int* __restrict__ dst,
                           const float* __restrict__ el, const float* __restrict__ er,
                           const unsigned* __restrict__ m, float* __restrict__ ex,
                           float* __restrict__ ssum, long nEH, int H) {
  long i = (long)blockIdx.x * blockDim.x + threadIdx.x;
  if (i >= nEH) return;
  int e = (int)(i / H), h = (int)(i % H);
  int s = src[e], d = dst[e];
  float v = lrelu(el[(size_t)s * H + h] + er[(size_t)d * H + h]);
  float x = expf(v - funkey(m[(size_t)d * H + h]));
  ex[i] = x;
  atomicAdd(&ssum[(size_t)d * H + h], x);
}

// acc[dst, t] += ex[e,h] * feat[src, t]   (t in [0, H*Dh), h = t/Dh)
__global__ void k_edge_agg(const int* __restrict__ src, const int* __restrict__ dst,
                           const float* __restrict__ feat, const float* __restrict__ ex,
                           float* __restrict__ acc, long nE, int H, int Dh) {
  int Dtot = H * Dh;
  long i = (long)blockIdx.x * blockDim.x + threadIdx.x;
  if (i >= nE * Dtot) return;
  int e = (int)(i / Dtot), t = (int)(i % Dtot);
  int h = t / Dh;
  int s = src[e], d = dst[e];
  atomicAdd(&acc[(size_t)d * Dtot + t],
            ex[(size_t)e * H + h] * feat[(size_t)s * Dtot + t]);
}

// layer-1 finalize: normalize, +b1, ELU, emit bf16 for the layer-2 GEMMs
__global__ void k_final1(const float* __restrict__ acc, const float* __restrict__ ssum,
                         const float* __restrict__ b, __bf16* __restrict__ h1b, long nN) {
  long i = (long)blockIdx.x * blockDim.x + threadIdx.x;
  if (i >= nN * D1) return;
  int n = (int)(i / D1), t = (int)(i % D1);
  float s = ssum[(size_t)n * HEADS1 + t / HIDDEN];
  float v = acc[i] / (s > 0.0f ? s : 1.0f) + b[t];
  v = v > 0.0f ? v : expf(v) - 1.0f;   // ELU
  h1b[i] = (__bf16)v;
}

// layer-2 finalize: normalize, +residual, +b2  (heads=1 -> mean over heads is identity)
__global__ void k_final2(const float* __restrict__ acc, const float* __restrict__ ssum,
                         const float* __restrict__ res, const float* __restrict__ b,
                         float* __restrict__ out, long nN) {
  long i = (long)blockIdx.x * blockDim.x + threadIdx.x;
  if (i >= nN * NCLS) return;
  int n = (int)(i / NCLS), t = (int)(i % NCLS);
  float s = ssum[n];
  out[i] = acc[i] / (s > 0.0f ? s : 1.0f) + res[i] + b[t];
}

// ---------------- host ----------------

static inline size_t align256(size_t x) { return (x + 255) & ~(size_t)255; }

extern "C" void kernel_launch(void* const* d_in, const int* in_sizes, int n_in,
                              void* d_out, int out_size, void* d_ws, size_t ws_size,
                              hipStream_t stream) {
  const float* x    = (const float*)d_in[0];
  const float* W1   = (const float*)d_in[1];
  const float* aL1  = (const float*)d_in[2];
  const float* aR1  = (const float*)d_in[3];
  const float* b1   = (const float*)d_in[4];
  const float* W2   = (const float*)d_in[5];
  const float* aL2  = (const float*)d_in[6];
  const float* aR2  = (const float*)d_in[7];
  const float* b2   = (const float*)d_in[8];
  const float* rW2  = (const float*)d_in[9];
  const int*   src  = (const int*)d_in[10];
  const int*   dst  = (const int*)d_in[11];
  float*       out  = (float*)d_out;

  const int  nN = in_sizes[0] / IN_DIM;   // 50000
  const long nE = in_sizes[10];           // 800000

  // workspace carve-out
  char* p = (char*)d_ws;
  size_t off = 0;
  auto take = [&](size_t bytes) { void* r = p + off; off += align256(bytes); return r; };

  __bf16*   xb    = (__bf16*)  take((size_t)nN * IN_DIM * 2);
  __bf16*   w1t   = (__bf16*)  take((size_t)D1 * IN_DIM * 2);     // [256 cols][256 K]
  __bf16*   w2t   = (__bf16*)  take((size_t)NCLS * D1 * 2);       // [64 cols][256 K]
  __bf16*   rwt   = (__bf16*)  take((size_t)NCLS * D1 * 2);
  float*    feat1 = (float*)   take((size_t)nN * D1 * 4);
  float*    el1   = (float*)   take((size_t)nN * HEADS1 * 4);
  float*    er1   = (float*)   take((size_t)nN * HEADS1 * 4);
  unsigned* m1    = (unsigned*)take((size_t)nN * HEADS1 * 4);
  float*    s1    = (float*)   take((size_t)nN * HEADS1 * 4);
  float*    ex1   = (float*)   take((size_t)nE * HEADS1 * 4);
  float*    acc1  = (float*)   take((size_t)nN * D1 * 4);
  __bf16*   h1b   = (__bf16*)  take((size_t)nN * D1 * 2);
  float*    feat2 = (float*)   take((size_t)nN * NCLS * 4);
  float*    res2  = (float*)   take((size_t)nN * NCLS * 4);
  float*    el2   = (float*)   take((size_t)nN * 4);
  float*    er2   = (float*)   take((size_t)nN * 4);
  unsigned* m2    = (unsigned*)take((size_t)nN * 4);
  float*    s2    = (float*)   take((size_t)nN * 4);
  float*    ex2   = (float*)   take((size_t)nE * 4);
  float*    acc2  = (float*)   take((size_t)nN * NCLS * 4);
  (void)ws_size;

  auto cdiv = [](long a, long b) { return (unsigned)((a + b - 1) / b); };

  // zero-init accumulators / maxima (fkey min == 0u)
  hipMemsetAsync(m1,   0, (size_t)nN * HEADS1 * 4, stream);
  hipMemsetAsync(s1,   0, (size_t)nN * HEADS1 * 4, stream);
  hipMemsetAsync(acc1, 0, (size_t)nN * D1 * 4, stream);
  hipMemsetAsync(m2,   0, (size_t)nN * 4, stream);
  hipMemsetAsync(s2,   0, (size_t)nN * 4, stream);
  hipMemsetAsync(acc2, 0, (size_t)nN * NCLS * 4, stream);

  // --- operand prep (bf16, weights transposed to K-contiguous columns) ---
  k_f32_to_bf16<<<cdiv((long)nN * IN_DIM, 256), 256, 0, stream>>>(x, xb, (long)nN * IN_DIM);
  k_transpose_bf16<<<cdiv(IN_DIM * D1, 256), 256, 0, stream>>>(W1,  w1t, IN_DIM, D1);
  k_transpose_bf16<<<cdiv(D1 * NCLS, 256), 256, 0, stream>>>(W2,  w2t, D1, NCLS);
  k_transpose_bf16<<<cdiv(D1 * NCLS, 256), 256, 0, stream>>>(rW2, rwt, D1, NCLS);

  // --- layer 1: feat1 = x @ W1  (WMMA bf16 -> f32) ---
  {
    long tiles = (long)(nN / 16) * (D1 / 16);       // 50000
    k_wmma_gemm<<<cdiv(tiles, 4), 128, 0, stream>>>(xb, w1t, feat1, nN, D1, IN_DIM);
  }
  k_eler<<<cdiv((long)nN * HEADS1, 256), 256, 0, stream>>>(feat1, aL1, aR1, el1, er1,
                                                           nN, HEADS1, HIDDEN);
  k_edge_max<<<cdiv(nE * HEADS1, 256), 256, 0, stream>>>(src, dst, el1, er1, m1,
                                                         nE * HEADS1, HEADS1);
  k_edge_exp<<<cdiv(nE * HEADS1, 256), 256, 0, stream>>>(src, dst, el1, er1, m1,
                                                         ex1, s1, nE * HEADS1, HEADS1);
  k_edge_agg<<<cdiv(nE * D1, 256), 256, 0, stream>>>(src, dst, feat1, ex1, acc1,
                                                     nE, HEADS1, HIDDEN);
  k_final1<<<cdiv((long)nN * D1, 256), 256, 0, stream>>>(acc1, s1, b1, h1b, nN);

  // --- layer 2: feat2 = h1 @ W2, res2 = h1 @ resW2 (fused dual WMMA) ---
  {
    long tiles = (long)(nN / 16) * (NCLS / 16);     // 12500
    k_wmma_gemm_dual<<<cdiv(tiles, 4), 128, 0, stream>>>(h1b, w2t, rwt, feat2, res2,
                                                         nN, NCLS, D1);
  }
  k_eler<<<cdiv((long)nN, 256), 256, 0, stream>>>(feat2, aL2, aR2, el2, er2, nN, 1, NCLS);
  k_edge_max<<<cdiv(nE, 256), 256, 0, stream>>>(src, dst, el2, er2, m2, nE, 1);
  k_edge_exp<<<cdiv(nE, 256), 256, 0, stream>>>(src, dst, el2, er2, m2, ex2, s2, nE, 1);
  k_edge_agg<<<cdiv(nE * NCLS, 256), 256, 0, stream>>>(src, dst, feat2, ex2, acc2,
                                                       nE, 1, NCLS);
  k_final2<<<cdiv((long)nN * NCLS, 256), 256, 0, stream>>>(acc2, s2, res2, b2, out, nN);
}
